// QuadLoRAModel_89335319757044
// MI455X (gfx1250) — matrix operand order repaired and checked by reading, not compiled, not run on hardware
//
#include <hip/hip_runtime.h>

typedef unsigned short u16;
typedef unsigned int   u32;
typedef __attribute__((ext_vector_type(16))) __bf16 v16bf;
typedef __attribute__((ext_vector_type(8)))  float  v8f;

// Problem dims (fixed by the reference)
constexpr int Bb = 4;     // batch
constexpr int S  = 4096;  // seq
constexpr int Dm = 2048;  // d_model (K of the GEMM)
constexpr int E  = 2048;  // d_out
constexpr int R  = 4;     // lora rank
constexpr int M  = 4;     // modes

constexpr int LDT   = 56;          // LDS row stride in u16 (112B: 16B aligned, conflict-free)
constexpr int BUFSZ = 128 * LDT;   // one LDS buffer, in u16 units
constexpr int NK    = Dm / 32;     // K-loop iterations

#if defined(__gfx1250__) && __has_builtin(__builtin_amdgcn_global_load_async_to_lds_b128)
#define HAVE_ASYNC 1
#else
#define HAVE_ASYNC 0
#endif

union FragBF { uint4 q[2]; v16bf v; };

#if HAVE_ASYNC
// Builtin signature (probe-confirmed): (global int4*, local int4*, imm offset, imm cpol)
typedef int vint4 __attribute__((vector_size(16)));
typedef __attribute__((address_space(1))) vint4 gvec_t;
typedef __attribute__((address_space(3))) vint4 lvec_t;

// One thread moves a 32B chunk of the x tile and a 32B chunk of the W tile.
// INST_OFFSET is added to BOTH the global and LDS address (ISA 08 §4.4).
__device__ __forceinline__ void async_tile(const u16* gx, const u16* gw,
                                           u16* lx, u16* lw) {
  __builtin_amdgcn_global_load_async_to_lds_b128((gvec_t*)gx, (lvec_t*)lx, 0, 0);
  __builtin_amdgcn_global_load_async_to_lds_b128((gvec_t*)gx, (lvec_t*)lx, 16, 0);
  __builtin_amdgcn_global_load_async_to_lds_b128((gvec_t*)gw, (lvec_t*)lw, 0, 0);
  __builtin_amdgcn_global_load_async_to_lds_b128((gvec_t*)gw, (lvec_t*)lw, 16, 0);
}

__device__ __forceinline__ void wg_sync() {          // raw split barrier, no counter waits
  asm volatile("" ::: "memory");
  __builtin_amdgcn_s_barrier();
  asm volatile("" ::: "memory");
}
__device__ __forceinline__ void wait_async_le4() {   // prev tile retired, next stays in flight
  asm volatile("s_wait_asynccnt 0x4" ::: "memory");
}
__device__ __forceinline__ void wait_async_0() {
  asm volatile("s_wait_asynccnt 0x0" ::: "memory");
}
#endif

// ---------------------------------------------------------------------------
// Kernel 0: x (f32) -> bf16 copy in workspace. 8 elements / thread.
// ---------------------------------------------------------------------------
__global__ __launch_bounds__(256) void convert_x(const float* __restrict__ x,
                                                 u16* __restrict__ xbf) {
  size_t i = ((size_t)blockIdx.x * 256 + threadIdx.x) * 8;
  float4 f0 = *(const float4*)(x + i);
  float4 f1 = *(const float4*)(x + i + 4);
  union { __bf16 h[8]; uint4 q; } u;
  u.h[0] = (__bf16)f0.x; u.h[1] = (__bf16)f0.y;
  u.h[2] = (__bf16)f0.z; u.h[3] = (__bf16)f0.w;
  u.h[4] = (__bf16)f1.x; u.h[5] = (__bf16)f1.y;
  u.h[6] = (__bf16)f1.z; u.h[7] = (__bf16)f1.w;
  *(uint4*)(xbf + i) = u.q;
}

// ---------------------------------------------------------------------------
// Kernel 1: fold base W + blended/scaled LoRA into per-batch bf16 W_eff[b][e][d]
//   W_eff[b,e,d] = W[e,d] + sum_{m,r} blend[b,m]*scales[m,b,r]*A[m,d,r]*Bm[m,r,e]
// ---------------------------------------------------------------------------
__global__ __launch_bounds__(256) void build_weff(
    const float* __restrict__ W, const float* __restrict__ A,
    const float* __restrict__ Bm, const float* __restrict__ scales,
    const float* __restrict__ blend, u16* __restrict__ weff) {
  u32 idx = blockIdx.x * 256u + threadIdx.x;
  u32 d4  = idx & (Dm / 4 - 1);          // 9 bits
  u32 e   = (idx >> 9) & (E - 1);        // 11 bits
  u32 b   = idx >> 20;                   // 2 bits

  float bmc[M][R];
  #pragma unroll
  for (int m = 0; m < M; ++m) {
    float bl = blend[b * M + m];
    #pragma unroll
    for (int r = 0; r < R; ++r)
      bmc[m][r] = bl * scales[(m * Bb + b) * R + r] * Bm[(m * R + r) * E + e];
  }

  int d0 = d4 * 4;
  float4 wv = *(const float4*)(W + (size_t)e * Dm + d0);
  float o[4] = {wv.x, wv.y, wv.z, wv.w};
  #pragma unroll
  for (int dd = 0; dd < 4; ++dd) {
    #pragma unroll
    for (int m = 0; m < M; ++m) {
      float4 av = *(const float4*)(A + ((size_t)m * Dm + d0 + dd) * R);
      o[dd] += av.x * bmc[m][0] + av.y * bmc[m][1] + av.z * bmc[m][2] + av.w * bmc[m][3];
    }
  }
  union { __bf16 h[4]; uint2 q; } st;
  st.h[0] = (__bf16)o[0]; st.h[1] = (__bf16)o[1];
  st.h[2] = (__bf16)o[2]; st.h[3] = (__bf16)o[3];
  *(uint2*)(weff + (size_t)(b * E + e) * Dm + d0) = st.q;
}

// ---------------------------------------------------------------------------
// Kernel 2: batched GEMM  out[b] (SxE) = xbf[b] (SxD) @ W_eff[b]^T, f32 accum.
// Block 128x128, 8 waves, wave tile 32x64 (2x4 WMMA accums).
// Tiles fed by async global->LDS (double buffered, 2 barriers/iter).
// ---------------------------------------------------------------------------
#define WMMA_ACC(ACC, AF, BF)                                                  \
  ACC = __builtin_amdgcn_wmma_f32_16x16x32_bf16(false, (AF), false, (BF),      \
                                                (short)0, (ACC), false, false)

__global__ __launch_bounds__(256) void qlora_gemm(
    const u16* __restrict__ xbf, const u16* __restrict__ weff,
    float* __restrict__ out) {
  __shared__ u16 sx[2 * BUFSZ];
  __shared__ u16 sw[2 * BUFSZ];

  const int tid  = threadIdx.x;
  const int lane = tid & 31;
  const int wv   = tid >> 5;
  const int wr   = wv >> 1;       // 0..3  (S direction, 32 rows each)
  const int wc   = wv & 1;        // 0..1  (E direction, 64 cols each)

  const int s0 = blockIdx.x * 128;
  const int e0 = blockIdx.y * 128;
  const int b  = blockIdx.z;

  const u16* xb = xbf  + (size_t)(b * S + s0) * Dm;
  const u16* wb = weff + (size_t)(b * E + e0) * Dm;

  // Tile copy assignment: thread -> (row, 16-element half-row) = 32B chunk
  const int lr = tid >> 1;         // 0..127
  const int lc = (tid & 1) * 16;   // 0 or 16 (u16 units)
  const u16* gx = xb + (size_t)lr * Dm + lc;
  const u16* gw = wb + (size_t)lr * Dm + lc;
  u16* lx = &sx[lr * LDT + lc];
  u16* lw = &sw[lr * LDT + lc];

  // Per-lane fragment geometry (CDNA5 WMMA VGPR layouts)
  const int arow = wr * 32 + (lane & 15);   // A: row M = lane&15
  const int ako  = (lane & 16) ? 8 : 0;     // A: K chunks at ako, ako+16
  const int brow = wc * 64 + (lane & 15);   // B: col N = lane&15 (= W_eff row e)
  const int bko  = (lane & 16) ? 8 : 0;     // B: K chunks at bko, bko+16

  // Named accumulators (no array indexing -> easier D/C coalescing for RA)
  v8f a00 = {}, a01 = {}, a02 = {}, a03 = {};
  v8f a10 = {}, a11 = {}, a12 = {}, a13 = {};

#if HAVE_ASYNC
  // ---- async double-buffered pipeline ----
  async_tile(gx, gw, lx, lw);               // tile 0 -> buffer 0
  int cur = 0;
  for (int kt = 0; kt < NK; ++kt) {
    const int  nxt  = cur ^ 1;
    const bool more = (kt + 1) < NK;

    wg_sync();                              // barrier A: WAR on buffer `nxt`
    if (more) {
      async_tile(gx + (kt + 1) * 32, gw + (kt + 1) * 32,
                 lx + nxt * BUFSZ, lw + nxt * BUFSZ);
      if (kt + 3 < NK) {                    // L2 prefetch a few tiles ahead
        __builtin_prefetch(gx + (kt + 3) * 32, 0, 1);
        __builtin_prefetch(gw + (kt + 3) * 32, 0, 1);
      }
      wait_async_le4();                     // tile kt done (in-order retire)
    } else {
      wait_async_0();
    }
    wg_sync();                              // barrier B: tile kt visible WG-wide

    const u16* cx = &sx[cur * BUFSZ];
    const u16* cw = &sw[cur * BUFSZ];
    FragBF af0, af1, bf0, bf1, bf2, bf3;
    {
      int ro = (arow + 0) * LDT;
      af0.q[0] = *(const uint4*)&cx[ro + ako];
      af0.q[1] = *(const uint4*)&cx[ro + ako + 16];
      ro = (arow + 16) * LDT;
      af1.q[0] = *(const uint4*)&cx[ro + ako];
      af1.q[1] = *(const uint4*)&cx[ro + ako + 16];
      ro = (brow + 0) * LDT;
      bf0.q[0] = *(const uint4*)&cw[ro + bko];
      bf0.q[1] = *(const uint4*)&cw[ro + bko + 16];
      ro = (brow + 16) * LDT;
      bf1.q[0] = *(const uint4*)&cw[ro + bko];
      bf1.q[1] = *(const uint4*)&cw[ro + bko + 16];
      ro = (brow + 32) * LDT;
      bf2.q[0] = *(const uint4*)&cw[ro + bko];
      bf2.q[1] = *(const uint4*)&cw[ro + bko + 16];
      ro = (brow + 48) * LDT;
      bf3.q[0] = *(const uint4*)&cw[ro + bko];
      bf3.q[1] = *(const uint4*)&cw[ro + bko + 16];
    }
    WMMA_ACC(a00, af0.v, bf0.v); WMMA_ACC(a01, af0.v, bf1.v);
    WMMA_ACC(a02, af0.v, bf2.v); WMMA_ACC(a03, af0.v, bf3.v);
    WMMA_ACC(a10, af1.v, bf0.v); WMMA_ACC(a11, af1.v, bf1.v);
    WMMA_ACC(a12, af1.v, bf2.v); WMMA_ACC(a13, af1.v, bf3.v);
    cur = nxt;
  }
#else
  // ---- fallback: register-staged double buffer, one barrier/iter ----
  {
    uint4 qx0 = *(const uint4*)gx, qx1 = *(const uint4*)(gx + 8);
    uint4 qw0 = *(const uint4*)gw, qw1 = *(const uint4*)(gw + 8);
    *(uint4*)lx = qx0; *(uint4*)(lx + 8) = qx1;
    *(uint4*)lw = qw0; *(uint4*)(lw + 8) = qw1;
  }
  int cur = 0;
  for (int kt = 0; kt < NK; ++kt) {
    __syncthreads();
    const bool more = (kt + 1) < NK;
    uint4 qx0, qx1, qw0, qw1;
    if (more) {
      const u16* g = gx + (kt + 1) * 32;
      const u16* h = gw + (kt + 1) * 32;
      qx0 = *(const uint4*)g; qx1 = *(const uint4*)(g + 8);
      qw0 = *(const uint4*)h; qw1 = *(const uint4*)(h + 8);
      if (kt + 3 < NK) {
        __builtin_prefetch(gx + (kt + 3) * 32, 0, 1);
        __builtin_prefetch(gw + (kt + 3) * 32, 0, 1);
      }
    }
    const u16* cx = &sx[cur * BUFSZ];
    const u16* cw = &sw[cur * BUFSZ];
    FragBF af0, af1, bf0, bf1, bf2, bf3;
    {
      int ro = (arow + 0) * LDT;
      af0.q[0] = *(const uint4*)&cx[ro + ako];
      af0.q[1] = *(const uint4*)&cx[ro + ako + 16];
      ro = (arow + 16) * LDT;
      af1.q[0] = *(const uint4*)&cx[ro + ako];
      af1.q[1] = *(const uint4*)&cx[ro + ako + 16];
      ro = (brow + 0) * LDT;
      bf0.q[0] = *(const uint4*)&cw[ro + bko];
      bf0.q[1] = *(const uint4*)&cw[ro + bko + 16];
      ro = (brow + 16) * LDT;
      bf1.q[0] = *(const uint4*)&cw[ro + bko];
      bf1.q[1] = *(const uint4*)&cw[ro + bko + 16];
      ro = (brow + 32) * LDT;
      bf2.q[0] = *(const uint4*)&cw[ro + bko];
      bf2.q[1] = *(const uint4*)&cw[ro + bko + 16];
      ro = (brow + 48) * LDT;
      bf3.q[0] = *(const uint4*)&cw[ro + bko];
      bf3.q[1] = *(const uint4*)&cw[ro + bko + 16];
    }
    WMMA_ACC(a00, af0.v, bf0.v); WMMA_ACC(a01, af0.v, bf1.v);
    WMMA_ACC(a02, af0.v, bf2.v); WMMA_ACC(a03, af0.v, bf3.v);
    WMMA_ACC(a10, af1.v, bf0.v); WMMA_ACC(a11, af1.v, bf1.v);
    WMMA_ACC(a12, af1.v, bf2.v); WMMA_ACC(a13, af1.v, bf3.v);
    if (more) {
      const int nxt = cur ^ 1;
      u16* px = lx + nxt * BUFSZ;
      u16* pw = lw + nxt * BUFSZ;
      *(uint4*)px = qx0; *(uint4*)(px + 8) = qx1;
      *(uint4*)pw = qw0; *(uint4*)(pw + 8) = qw1;
    }
    cur ^= 1;
  }
#endif

  // Epilogue: C/D layout -> coalesced f32 stores
  // lanes 0-15: N=lane, M=v ; lanes 16-31: N=lane-16, M=v+8
  const int srow0 = s0 + wr * 32 + ((lane & 16) >> 1);
  const int ecol0 = e0 + wc * 64 + (lane & 15);
#define STORE_ACC(ACC, I, J)                                                   \
  _Pragma("unroll")                                                            \
  for (int v = 0; v < 8; ++v) {                                                \
    out[(size_t)(b * S + (srow0 + (I) * 16 + v)) * E + (ecol0 + (J) * 16)] =   \
        (ACC)[v];                                                              \
  }
  STORE_ACC(a00, 0, 0) STORE_ACC(a01, 0, 1) STORE_ACC(a02, 0, 2) STORE_ACC(a03, 0, 3)
  STORE_ACC(a10, 1, 0) STORE_ACC(a11, 1, 1) STORE_ACC(a12, 1, 2) STORE_ACC(a13, 1, 3)
#undef STORE_ACC
}

// ---------------------------------------------------------------------------
extern "C" void kernel_launch(void* const* d_in, const int* in_sizes, int n_in,
                              void* d_out, int out_size, void* d_ws, size_t ws_size,
                              hipStream_t stream) {
  (void)in_sizes; (void)n_in; (void)out_size; (void)ws_size;
  const float* x      = (const float*)d_in[0];
  const float* W      = (const float*)d_in[1];
  const float* A      = (const float*)d_in[2];
  const float* Bm     = (const float*)d_in[3];
  const float* scales = (const float*)d_in[4];
  const float* blend  = (const float*)d_in[5];

  // Workspace layout: [0, 32MB) W_eff bf16 ; [32MB, 96MB) x bf16
  u16* weff = (u16*)d_ws;
  u16* xbf  = (u16*)d_ws + (size_t)Bb * E * Dm;

  convert_x<<<(Bb * S * Dm / 8) / 256, 256, 0, stream>>>(x, xbf);

  build_weff<<<(Bb * E * (Dm / 4)) / 256, 256, 0, stream>>>(W, A, Bm, scales,
                                                            blend, weff);

  dim3 grid(S / 128, E / 128, Bb);
  qlora_gemm<<<grid, 256, 0, stream>>>(xbf, weff, (float*)d_out);
}
